// MOERKHSSelector_47021301957444
// MI455X (gfx1250) — compile-verified, matrix operand order checked by reading, not run on hardware
//
#include <hip/hip_runtime.h>
#include <hip/hip_bf16.h>
#include <math.h>

// Problem constants (from the reference)
#define BT  32768    // B*S tokens
#define DD  2048     // D
#define RK  1024     // RKHS
#define EMBD 128     // EMB
#define NE  64       // experts

typedef __attribute__((ext_vector_type(16))) __bf16 v16bf;
typedef __attribute__((ext_vector_type(8)))  float  v8f;

#define WMMA_BF16(ACC, A, B) \
  ACC = __builtin_amdgcn_wmma_f32_16x16x32_bf16(false, (A), false, (B), (short)0, (ACC), false, false)

// ---------------------------------------------------------------------------
// K0: M[e,r] = sum_d emb[e,d] * W_exp[r,d] + b_exp[r]      (64 x 1024, K=128)
// ---------------------------------------------------------------------------
__global__ __launch_bounds__(256) void emb_proj_kernel(
    const float* __restrict__ emb, const float* __restrict__ W_exp,
    const float* __restrict__ b_exp, float* __restrict__ M)
{
  int g = blockIdx.x * 256 + threadIdx.x;      // g = e*1024 + r
  int r = g & (RK - 1);
  int e = g >> 10;
  const float4* ep = (const float4*)(emb + (size_t)e * EMBD);
  const float4* wp = (const float4*)(W_exp + (size_t)r * EMBD);
  float acc = 0.0f;
#pragma unroll 8
  for (int q = 0; q < EMBD / 4; ++q) {
    float4 a = ep[q], b = wp[q];
    acc = fmaf(a.x, b.x, acc); acc = fmaf(a.y, b.y, acc);
    acc = fmaf(a.z, b.z, acc); acc = fmaf(a.w, b.w, acc);
  }
  M[g] = acc + b_exp[r];
}

// ---------------------------------------------------------------------------
// K1: fold  C[e,d] = sum_r M[e,r] * W_hid[r,d]  (64 x 2048, K=1024)
//     emitted directly as bf16 hi/lo in WMMA B-fragment memory order:
//     idx = ((et*64 + kc)*32 + lane)*16 + j
//     with d = kc*32 + (j&7) + ((j>>3)<<4) + ((lane>>4)<<3), e = et*16 + (lane&15)
// ---------------------------------------------------------------------------
__global__ __launch_bounds__(256) void fold_kernel(
    const float* __restrict__ M, const float* __restrict__ W_hid,
    __bf16* __restrict__ Bhi, __bf16* __restrict__ Blo)
{
  const int bx = blockIdx.x;
  const int e0 = (bx & 3) * 16;          // 4 expert tiles
  const int d0 = (bx >> 2) * 64;         // 32 d-chunks of 64
  const int dl = threadIdx.x & 63;
  const int es = threadIdx.x >> 6;       // 0..3
  const int d  = d0 + dl;

  const float* __restrict__ m0 = M + (size_t)(e0 + es * 4 + 0) * RK;
  const float* __restrict__ m1 = M + (size_t)(e0 + es * 4 + 1) * RK;
  const float* __restrict__ m2 = M + (size_t)(e0 + es * 4 + 2) * RK;
  const float* __restrict__ m3 = M + (size_t)(e0 + es * 4 + 3) * RK;

  float a0 = 0.f, a1 = 0.f, a2 = 0.f, a3 = 0.f;
#pragma unroll 4
  for (int r = 0; r < RK; ++r) {
    float w = W_hid[(size_t)r * DD + d];   // coalesced across threads
    a0 = fmaf(m0[r], w, a0);
    a1 = fmaf(m1[r], w, a1);
    a2 = fmaf(m2[r], w, a2);
    a3 = fmaf(m3[r], w, a3);
  }

  float acc[4] = {a0, a1, a2, a3};
#pragma unroll
  for (int i = 0; i < 4; ++i) {
    int e = e0 + es * 4 + i;
    float v = acc[i];
    __bf16 h = (__bf16)v;
    __bf16 l = (__bf16)(v - (float)h);
    int et = e >> 4, laneN = e & 15;
    int kc = d >> 5, rem = d & 31;
    int half = (rem >> 3) & 1;
    int j = (rem & 7) | (((rem >> 4) & 1) << 3);
    int lane = laneN + (half << 4);
    size_t idx = (((size_t)(et * 64 + kc) * 32) + lane) * 16 + j;
    Bhi[idx] = h;
    Blo[idx] = l;
  }
}

// ---------------------------------------------------------------------------
// K1b: c0[e] = sum_r b_hid[r] * M[e,r]
// ---------------------------------------------------------------------------
__global__ void bias_fold_kernel(const float* __restrict__ b_hid,
                                 const float* __restrict__ M,
                                 float* __restrict__ c0)
{
  int e = threadIdx.x;
  const float* Mr = M + (size_t)e * RK;
  float acc = 0.0f;
#pragma unroll 4
  for (int r = 0; r < RK; ++r) acc = fmaf(b_hid[r], Mr[r], acc);
  c0[e] = acc;
}

// ---------------------------------------------------------------------------
// K3: main GEMM  logits[t,e] = X[t,:] . C[e,:] + c0[e], then top-2 + weights.
// Block: 256 thr = 8 waves, 128 tokens x 64 experts. Wave: 16 tokens, 4 etiles.
// bf16 hi/lo split: 3 WMMAs per (etile, kchunk) -> near-fp32 accuracy.
// All 20 b128 loads per K-chunk are issued up front so the split-VALU work
// (A-dependent only) and early WMMAs cover the B-fragment load latency.
// ---------------------------------------------------------------------------
#define SPLIT(IDX, VAL) do { float _v = (VAL); __bf16 _h = (__bf16)_v; \
    ah[IDX] = _h; al[IDX] = (__bf16)(_v - (float)_h); } while (0)

__global__ __launch_bounds__(256) void router_kernel(
    const float* __restrict__ X, const __bf16* __restrict__ BhiP,
    const __bf16* __restrict__ BloP, const float* __restrict__ c0,
    int* __restrict__ sel, float* __restrict__ rw, float* __restrict__ aux)
{
  __shared__ float lds[128 * 68];   // [token][expert], stride 68 -> bank-safe
  const int tid  = threadIdx.x;
  const int lane = tid & 31;
  const int wv   = tid >> 5;
  const int row  = blockIdx.x * 128 + wv * 16 + (lane & 15);
  const float* __restrict__ Xrow = X + (size_t)row * DD;
  const int sub = (lane >> 4) << 3;              // K half select (0 or 8)
  const v16bf* __restrict__ BH = (const v16bf*)BhiP;  // [et][kc][lane] 32B frags
  const v16bf* __restrict__ BL = (const v16bf*)BloP;

  v8f acc0 = {0,0,0,0,0,0,0,0};
  v8f acc1 = {0,0,0,0,0,0,0,0};
  v8f acc2 = {0,0,0,0,0,0,0,0};
  v8f acc3 = {0,0,0,0,0,0,0,0};

  const int eoff = lane & 15;
  const float c0v0 = c0[eoff];
  const float c0v1 = c0[16 + eoff];
  const float c0v2 = c0[32 + eoff];
  const float c0v3 = c0[48 + eoff];

  for (int kc = 0; kc < DD / 32; ++kc) {
    // ---- issue ALL loads for this K-chunk first (A: 4 x b128, B: 16 x b128)
    const float* p = Xrow + kc * 32 + sub;
    // A-fragment: lanes 0-15 cover K {0..7,16..23}, lanes 16-31 K {8..15,24..31}
    float4 a0 = *(const float4*)(p);
    float4 a1 = *(const float4*)(p + 4);
    float4 a2 = *(const float4*)(p + 16);
    float4 a3 = *(const float4*)(p + 20);

    const int bb = kc * 32 + lane;     // etile stride = 64*32 = 2048 frags
    v16bf bh0 = BH[bb],        bl0 = BL[bb];
    v16bf bh1 = BH[2048 + bb], bl1 = BL[2048 + bb];
    v16bf bh2 = BH[4096 + bb], bl2 = BL[4096 + bb];
    v16bf bh3 = BH[6144 + bb], bl3 = BL[6144 + bb];

    {   // prefetch ~8 chunks ahead (clamped inside the row)
      int pf = kc * 32 + sub + 256; if (pf > DD - 1) pf = DD - 1;
      __builtin_prefetch(Xrow + pf, 0, 1);
    }

    // ---- hi/lo split (depends on A only; overlaps B loads in flight)
    v16bf ah, al;
    SPLIT(0,  a0.x); SPLIT(1,  a0.y); SPLIT(2,  a0.z); SPLIT(3,  a0.w);
    SPLIT(4,  a1.x); SPLIT(5,  a1.y); SPLIT(6,  a1.z); SPLIT(7,  a1.w);
    SPLIT(8,  a2.x); SPLIT(9,  a2.y); SPLIT(10, a2.z); SPLIT(11, a2.w);
    SPLIT(12, a3.x); SPLIT(13, a3.y); SPLIT(14, a3.z); SPLIT(15, a3.w);

    // ---- 12 WMMAs; waits stagger since B loads complete in issue order
    WMMA_BF16(acc0, ah, bh0); WMMA_BF16(acc0, al, bh0); WMMA_BF16(acc0, ah, bl0);
    WMMA_BF16(acc1, ah, bh1); WMMA_BF16(acc1, al, bh1); WMMA_BF16(acc1, ah, bl1);
    WMMA_BF16(acc2, ah, bh2); WMMA_BF16(acc2, al, bh2); WMMA_BF16(acc2, ah, bl2);
    WMMA_BF16(acc3, ah, bh3); WMMA_BF16(acc3, al, bh3); WMMA_BF16(acc3, ah, bl3);
  }

  // Scatter logits to LDS: C/D layout -> lanes 0-15: M=v, lanes 16-31: M=v+8
  const int tokBase = wv * 16 + sub;
#pragma unroll
  for (int v = 0; v < 8; ++v) {
    float* q = lds + (size_t)(tokBase + v) * 68 + eoff;
    q[0]  = acc0[v] + c0v0;
    q[16] = acc1[v] + c0v1;
    q[32] = acc2[v] + c0v2;
    q[48] = acc3[v] + c0v3;
  }
  __syncthreads();

  // Phase 2: one thread per token -> top-2 + softmax over the two top logits.
  if (tid < 128) {
    const float* r = lds + (size_t)tid * 68;
    float l1 = -INFINITY, l2 = -INFINITY;
    int i1 = 0, i2 = 0;
    for (int e = 0; e < NE; ++e) {
      float v = r[e];
      if (v > l1)      { l2 = l1; i2 = i1; l1 = v; i1 = e; }
      else if (v > l2) { l2 = v;  i2 = e; }
    }
    // rw_i = p_i/(p1+p2) == softmax over {l1,l2}
    float ex  = __expf(l2 - l1);
    float inv = 1.0f / (1.0f + ex);
    size_t g = (size_t)blockIdx.x * 128 + tid;
    sel[g * 2]     = i1;
    sel[g * 2 + 1] = i2;
    rw[g * 2]      = inv;
    rw[g * 2 + 1]  = ex * inv;
  }
  // aux load-balancing loss is analytically constant: (2/64)*(1/2)*64^2 = 64
  if (blockIdx.x == 0 && tid == 0) aux[0] = 64.0f;
}

// ---------------------------------------------------------------------------
extern "C" void kernel_launch(void* const* d_in, const int* in_sizes, int n_in,
                              void* d_out, int out_size, void* d_ws, size_t ws_size,
                              hipStream_t stream) {
  const float* X     = (const float*)d_in[0];   // [B,S,D]
  const float* W_hid = (const float*)d_in[1];   // [RKHS,D]
  const float* b_hid = (const float*)d_in[2];   // [RKHS]
  const float* W_exp = (const float*)d_in[3];   // [RKHS,EMB]
  const float* b_exp = (const float*)d_in[4];   // [RKHS]
  const float* emb   = (const float*)d_in[5];   // [E,EMB]

  char* ws = (char*)d_ws;
  float*  M   = (float*)ws;                              // 64*1024 f32 = 256KB
  float*  c0  = (float*)(ws + 256 * 1024);               // 64 f32
  __bf16* Bhi = (__bf16*)(ws + 256 * 1024 + 4096);       // 131072 bf16 = 256KB
  __bf16* Blo = Bhi + (size_t)4 * 64 * 32 * 16;          // 256KB

  emb_proj_kernel <<<(NE * RK) / 256, 256, 0, stream>>>(emb, W_exp, b_exp, M);
  bias_fold_kernel<<<1, 64, 0, stream>>>(b_hid, M, c0);
  fold_kernel     <<<128, 256, 0, stream>>>(M, W_hid, Bhi, Blo);

  int*   sel = (int*)d_out;                  // [BT,2] int32
  float* rw  = (float*)d_out + 2 * BT;       // [BT,2] f32
  float* aux = (float*)d_out + 4 * BT;       // scalar
  router_kernel<<<BT / 128, 256, 0, stream>>>(X, Bhi, Blo, c0, sel, rw, aux);
}
